// SingleHeadSelfAttention_14319420964867
// MI455X (gfx1250) — matrix operand order repaired and checked by reading, not compiled, not run on hardware
//
#include <hip/hip_runtime.h>
#include <hip/hip_bf16.h>

typedef __bf16 bf16;
typedef __attribute__((ext_vector_type(16))) __bf16 v16bf;
typedef __attribute__((ext_vector_type(8)))  __bf16 v8bf;
typedef __attribute__((ext_vector_type(4)))  __bf16 v4bf;
typedef __attribute__((ext_vector_type(8)))  float  v8f;
typedef unsigned int u32;
typedef __attribute__((ext_vector_type(4)))  u32 u32x4;
typedef __attribute__((ext_vector_type(8)))  u32 u32x8;

#define B_ 4
#define S_ 2048
#define E_ 1024
#define H_ 512

// Build a 16-element bf16 fragment from two contiguous 16-byte loads.
__device__ __forceinline__ v16bf ld_frag(const bf16* p0, const bf16* p1) {
  union { v16bf v; v8bf h[2]; } u;
  u.h[0] = *(const v8bf*)p0;
  u.h[1] = *(const v8bf*)p1;
  return u.v;
}

__device__ __forceinline__ v8f wmma_bf16(v16bf a, v16bf b, v8f c) {
  // D(f32 16x16) = A(16x32 bf16) * B(32x16 bf16) + C
  return __builtin_amdgcn_wmma_f32_16x16x32_bf16(
      /*neg_a=*/false, a, /*neg_b=*/false, b,
      /*c_mod=*/(short)0, c, /*reuse_a=*/false, /*reuse_b=*/false);
}

// ---------------- Stage 0a: fp32 -> bf16 elementwise ----------------
__global__ void cvt_bf16_kernel(const float* __restrict__ in,
                                bf16* __restrict__ out, int n4) {
  int i = blockIdx.x * blockDim.x + threadIdx.x;
  if (i >= n4) return;
  const float4 f = *(const float4*)(in + (size_t)i * 4);
  v4bf o;
  o[0] = (bf16)f.x; o[1] = (bf16)f.y; o[2] = (bf16)f.z; o[3] = (bf16)f.w;
  *(v4bf*)(out + (size_t)i * 4) = o;
}

// ---------------- Stage 0b: W [E,H] fp32 -> Wt [3][H][E] bf16 ----------------
__global__ void wt_kernel(const float* __restrict__ Wq,
                          const float* __restrict__ Wk,
                          const float* __restrict__ Wv,
                          bf16* __restrict__ Wt) {
  int idx = blockIdx.x * blockDim.x + threadIdx.x;
  if (idx >= 3 * E_ * H_) return;
  int mat = idx / (E_ * H_);
  int r   = idx % (E_ * H_);
  int e = r / H_, h = r % H_;
  const float* W = (mat == 0) ? Wq : ((mat == 1) ? Wk : Wv);
  Wt[(size_t)mat * H_ * E_ + (size_t)h * E_ + e] = (bf16)W[r];
}

// ---------------- Stage 1: fused QKV projection GEMM ----------------
// One wave per 16x64 output tile. Q,K -> row-major [B*S][H] bf16.
// V -> transposed [B][H][S] bf16 (so P@V B-fragments are contiguous).
__global__ __launch_bounds__(256) void proj_kernel(
    const bf16* __restrict__ qbf, const bf16* __restrict__ kvbf,
    const bf16* __restrict__ Wt,
    bf16* __restrict__ Qh, bf16* __restrict__ Kh, bf16* __restrict__ Vt) {
  const int lane = threadIdx.x & 31;
  const int wid  = blockIdx.x * 8 + (threadIdx.x >> 5);
  const int TILES = (B_ * S_ / 16) * (H_ / 64);   // 4096 per matrix
  const int mat = wid / TILES;                    // 0=Q,1=K,2=V
  const int r   = wid % TILES;
  const int mtile = r / (H_ / 64);
  const int nt64  = r % (H_ / 64);
  const bf16* A = (mat == 0) ? qbf : kvbf;
  const bf16* W = Wt + (size_t)mat * H_ * E_;
  const int mbase = mtile * 16;
  const int nbase = nt64 * 64;
  const int m = lane & 15, sel = lane >> 4;

  v8f acc[4] = {};
  const bf16* arow = A + (size_t)(mbase + m) * E_;
  for (int e = 0; e < E_; e += 32) {
    // A fragment: row m, 16 contiguous K starting e + sel*16
    v16bf af = ld_frag(arow + e + sel * 16, arow + e + sel * 16 + 8);
#pragma unroll
    for (int t = 0; t < 4; ++t) {
      // B fragment: col n = m, 16 contiguous K starting e + sel*16
      const bf16* wrow = W + (size_t)(nbase + t * 16 + m) * E_ + e + sel * 16;
      v16bf bfr = ld_frag(wrow, wrow + 8);
      acc[t] = wmma_bf16(af, bfr, acc[t]);
    }
  }

  if (mat < 2) {
    bf16* O = (mat == 0) ? Qh : Kh;
#pragma unroll
    for (int t = 0; t < 4; ++t)
#pragma unroll
      for (int v = 0; v < 8; ++v) {
        int row = mbase + v + 8 * sel;        // C layout: M = v + 8*(lane/16)
        int col = nbase + t * 16 + m;         //           N = lane%16
        O[(size_t)row * H_ + col] = (bf16)acc[t][v];
      }
  } else {
    // Vt[b][h][s]: consecutive v = consecutive s -> one 16B store per tile
#pragma unroll
    for (int t = 0; t < 4; ++t) {
      int col  = nbase + t * 16 + m;          // h
      int row0 = mbase + 8 * sel;             // s base (multiple of 8)
      int b = row0 / S_, s = row0 % S_;
      v8bf pk;
#pragma unroll
      for (int v = 0; v < 8; ++v) pk[v] = (bf16)acc[t][v];
      *(v8bf*)(Vt + ((size_t)b * H_ + col) * S_ + s) = pk;
    }
  }
}

// ---------------- Stage 2: causal flash attention ----------------
// Block = 8 waves = 4 q-tiles x 2 h-blocks of 256. Online softmax per wave.
// Q tiles staged in LDS once via the Tensor Data Mover (one contiguous 64KB
// tile), shared by the 2 h-block waves of each q-tile.
__global__ __launch_bounds__(256) void attn_kernel(
    const bf16* __restrict__ Qh, const bf16* __restrict__ Kh,
    const bf16* __restrict__ Vt, float* __restrict__ out) {
  __shared__ bf16 Qlds[4][16 * H_];   // 64 KB: four Q tiles, row-major
  __shared__ bf16 Plds[8][16 * 32];   // 8 KB : per-wave P staging

  const int tid  = threadIdx.x;
  const int lane = tid & 31;
  const int w    = tid >> 5;
  const int qt0  = blockIdx.x * 4;

  // ---- TDM: one 1-D tensor_load_to_lds moves all 4 Q tiles (64KB). ----
  if (w == 0) {
    const unsigned long long ga =
        (unsigned long long)(uintptr_t)(Qh + (size_t)qt0 * 16 * H_);
    const u32 lds_addr = (u32)(uintptr_t)(&Qlds[0][0]);  // wave-relative LDS byte addr
    const u32 NE = (4 * 16 * H_ * 2) / 8;                // 8192 elements of 8 bytes
    u32x4 g0;
    g0[0] = 1u;                                   // count=1 valid descriptor
    g0[1] = lds_addr;                             // lds_addr [63:32]
    g0[2] = (u32)ga;                              // global_addr [95:64]
    g0[3] = (u32)(ga >> 32) | (2u << 30);         // global_addr hi | type=2
    u32x8 g1;
    g1[0] = 3u << 16;                             // workgroup_mask=0, data_size=8B
    g1[1] = (NE & 0xffffu) << 16;                 // tensor_dim0[15:0]  (bits 63:48)
    g1[2] = (NE >> 16) & 0xffffu;                 // tensor_dim0[31:16] (bits 79:64)
    g1[3] = (NE & 0xffffu) << 16;                 // tile_dim0          (bits 127:112)
    g1[4] = 0u;                                   // tile_dim1=0, tile_dim2=0 (1-D)
    g1[5] = NE;                                   // tensor_dim0_stride[31:0]
    g1[6] = 0u;                                   // stride hi, dim1_stride lo
    g1[7] = 0u;
    asm volatile("tensor_load_to_lds %0, %1" :: "s"(g0), "s"(g1) : "memory");
    __builtin_amdgcn_s_wait_tensorcnt(0);
  }
  __syncthreads();

  const int qt    = qt0 + (w >> 1);
  const int hb    = (w & 1) * 256;
  const int b     = (qt * 16) / S_;
  const int qbase = (qt * 16) % S_;
  const bf16* Qt  = Qlds[w >> 1];
  const bf16* KbP = Kh + (size_t)b * S_ * H_;
  const bf16* VbP = Vt + (size_t)b * H_ * S_;
  const int m = lane & 15, sel = lane >> 4;
  const float scale = rsqrtf((float)H_);

  v8f O[16] = {};                       // 16 queries x 256 h columns, f32
  float mrow[8], lrow[8];
#pragma unroll
  for (int v = 0; v < 8; ++v) { mrow[v] = -1e30f; lrow[v] = 0.f; }

  const int ksteps = (qbase + 16 + 31) >> 5;   // causal: keys 0..qbase+15
  for (int ks = 0; ks < ksteps; ++ks) {
    const int kb = ks * 32;
    // Prefetch next key step's K rows toward the WGP while we compute.
    if (ks + 1 < ksteps)
      __builtin_prefetch(KbP + (size_t)(kb + 32 + lane) * H_, 0, 1);

    v8f s0 = {}, s1 = {};
    // Scores: 16x32 tile of Q.K^T, reduced over full H=512
    for (int e = 0; e < H_; e += 32) {
      v16bf qa = ld_frag(Qt + m * H_ + e + sel * 16,
                         Qt + m * H_ + e + sel * 16 + 8);
      const bf16* k0p = KbP + (size_t)(kb + m) * H_ + e + sel * 16;
      s0 = wmma_bf16(qa, ld_frag(k0p, k0p + 8), s0);
      const bf16* k1p = KbP + (size_t)(kb + 16 + m) * H_ + e + sel * 16;
      s1 = wmma_bf16(qa, ld_frag(k1p, k1p + 8), s1);
    }
    // Mask + scale + online softmax stats (wave32 butterfly in 16-lane halves)
    float p0[8], p1[8], alpha[8];
#pragma unroll
    for (int v = 0; v < 8; ++v) {
      const int qrow = qbase + v + 8 * sel;
      float a = s0[v] * scale, c = s1[v] * scale;
      if (kb + m      > qrow) a = -1e30f;
      if (kb + 16 + m > qrow) c = -1e30f;
      float x = fmaxf(a, c);
      x = fmaxf(x, __shfl_xor(x, 1, 32));
      x = fmaxf(x, __shfl_xor(x, 2, 32));
      x = fmaxf(x, __shfl_xor(x, 4, 32));
      x = fmaxf(x, __shfl_xor(x, 8, 32));
      const float mnew = fmaxf(mrow[v], x);
      alpha[v] = __expf(mrow[v] - mnew);
      mrow[v]  = mnew;
      p0[v] = __expf(a - mnew);
      p1[v] = __expf(c - mnew);
      float rsum = p0[v] + p1[v];
      rsum += __shfl_xor(rsum, 1, 32);
      rsum += __shfl_xor(rsum, 2, 32);
      rsum += __shfl_xor(rsum, 4, 32);
      rsum += __shfl_xor(rsum, 8, 32);
      lrow[v] = lrow[v] * alpha[v] + rsum;
    }
    // Rescale running output
#pragma unroll
    for (int t = 0; t < 16; ++t)
#pragma unroll
      for (int v = 0; v < 8; ++v) O[t][v] *= alpha[v];
    // C-layout -> A-layout for P via wave-private LDS round trip
    bf16* P = &Plds[w][0];
#pragma unroll
    for (int v = 0; v < 8; ++v) {
      const int row = v + 8 * sel;
      P[row * 32 + m]      = (bf16)p0[v];
      P[row * 32 + 16 + m] = (bf16)p1[v];
    }
    asm volatile("s_wait_dscnt 0" ::: "memory");  // LDS RAW within wave
    v16bf pa = ld_frag(P + m * 32 + sel * 16, P + m * 32 + sel * 16 + 8);
    // O += P @ V  (V transposed: contiguous 16-key runs per h column)
#pragma unroll
    for (int t = 0; t < 16; ++t) {
      const bf16* vp = VbP + (size_t)(hb + t * 16 + m) * S_ + kb + sel * 16;
      O[t] = wmma_bf16(pa, ld_frag(vp, vp + 8), O[t]);
    }
  }
  // Normalize and store fp32 output [B*S][H]
#pragma unroll
  for (int v = 0; v < 8; ++v) lrow[v] = 1.0f / lrow[v];
  float* outp = out + (size_t)qt * 16 * H_;
#pragma unroll
  for (int t = 0; t < 16; ++t)
#pragma unroll
    for (int v = 0; v < 8; ++v)
      outp[(size_t)(v + 8 * sel) * H_ + hb + t * 16 + m] = O[t][v] * lrow[v];
}

extern "C" void kernel_launch(void* const* d_in, const int* in_sizes, int n_in,
                              void* d_out, int out_size, void* d_ws, size_t ws_size,
                              hipStream_t stream) {
  const float* q   = (const float*)d_in[0];
  const float* k_v = (const float*)d_in[1];
  const float* Wq  = (const float*)d_in[2];
  const float* Wk  = (const float*)d_in[3];
  const float* Wv  = (const float*)d_in[4];
  // d_in[5] = boolean mask: unused, causal mask is applied analytically.

  // Workspace layout (bf16 elements)
  bf16* ws = (bf16*)d_ws;
  size_t off = 0;
  bf16* qbf  = ws + off; off += (size_t)B_ * S_ * E_;   // 8.4M
  bf16* kvbf = ws + off; off += (size_t)B_ * S_ * E_;   // 8.4M
  bf16* Wt   = ws + off; off += (size_t)3 * H_ * E_;    // 1.6M
  bf16* Qh   = ws + off; off += (size_t)B_ * S_ * H_;   // 4.2M
  bf16* Kh   = ws + off; off += (size_t)B_ * S_ * H_;   // 4.2M
  bf16* Vt   = ws + off; off += (size_t)B_ * S_ * H_;   // 4.2M  (~59 MB total)

  const int nQ  = B_ * S_ * E_;
  cvt_bf16_kernel<<<(nQ / 4 + 255) / 256, 256, 0, stream>>>(q,   qbf,  nQ / 4);
  cvt_bf16_kernel<<<(nQ / 4 + 255) / 256, 256, 0, stream>>>(k_v, kvbf, nQ / 4);

  const int nW = 3 * E_ * H_;
  wt_kernel<<<(nW + 255) / 256, 256, 0, stream>>>(Wq, Wk, Wv, Wt);

  // 3 matrices * 4096 tiles = 12288 waves = 1536 blocks of 8 waves
  proj_kernel<<<1536, 256, 0, stream>>>(qbf, kvbf, Wt, Qh, Kh, Vt);

  // 512 q-tiles / 4 per block = 128 blocks
  attn_kernel<<<128, 256, 0, stream>>>(Qh, Kh, Vt, (float*)d_out);
}
